// CompetitiveSparse_70068096467281
// MI455X (gfx1250) — matrix operand order repaired and checked by reading, not compiled, not run on hardware
//
#include <hip/hip_runtime.h>
#include <hip/hip_bf16.h>
#include <math.h>

// ---------------------------------------------------------------------------
// Problem constants (match reference)
// ---------------------------------------------------------------------------
#define BATCH 4096
#define DIM   4096
#define THRESHOLD 0.5f
#define COMP_FACTOR 0.8f

// GEMM tiling
#define BM 128
#define BN 128
#define BK 32

typedef __attribute__((ext_vector_type(16))) __bf16 v16bf;
typedef __attribute__((ext_vector_type(8)))  float  v8f;

union FragU {
    v16bf bf;
    uint4 q[2];
};

// Pack two f32 into one dword of bf16 (lo -> [15:0], hi -> [31:16]).
// Round-half-up via +0x8000 on the bit pattern, then a single v_perm_b32
// merges the two high halves: 3 VALU ops per pair.
__device__ __forceinline__ unsigned pack_bf16(float lo, float hi) {
    unsigned a = __float_as_uint(lo) + 0x8000u;
    unsigned b = __float_as_uint(hi) + 0x8000u;
#if __has_builtin(__builtin_amdgcn_perm)
    return __builtin_amdgcn_perm(b, a, 0x07060302u);  // {b.b3,b.b2,a.b3,a.b2}
#else
    return (a >> 16) | (b & 0xffff0000u);
#endif
}

// ---------------------------------------------------------------------------
// Kernel 0: zero the per-row accumulator (we must self-initialize workspace)
// ---------------------------------------------------------------------------
__global__ void zero_acc_kernel(float* __restrict__ acc, int n) {
    int i = blockIdx.x * blockDim.x + threadIdx.x;
    if (i < n) acc[i] = 0.0f;
}

// ---------------------------------------------------------------------------
// Kernel 1: fused  acc[b] += sum_n relu( (F @ W1)[b,n] + b1[n] ) * W2[n]
// 128x128 tile / workgroup, BK=32, double-buffered LDS, bf16 WMMA f32-accum.
// 256 threads = 8 wave32; wave (mw,nw) owns a 64x32 patch = 4x2 WMMA tiles.
// ---------------------------------------------------------------------------
__global__ __launch_bounds__(256)
void gemm_comp_kernel(const float* __restrict__ F,
                      const float* __restrict__ W1,
                      const float* __restrict__ b1,
                      const float* __restrict__ W2,
                      float* __restrict__ acc)
{
    __shared__ __align__(16) unsigned short lds_a[2][BM * BK];  // [m][k] bf16
    __shared__ __align__(16) unsigned short lds_b[2][BN * BK];  // [n][k] bf16
    __shared__ float acc_lds[BM];

    const int t    = threadIdx.x;
    const int lane = t & 31;
    const int w    = t >> 5;
    const int mw   = w >> 2;        // 0..1  -> 64-row slab
    const int nw   = w & 3;         // 0..3  -> 32-col slab
    const int row0 = blockIdx.y * BM;
    const int col0 = blockIdx.x * BN;

    v8f acc_c[4][2] = {};           // 8 x (16x16 f32) accumulators per wave

    // A staging: thread -> (row ar, 16-wide half ah)
    const int ar = t >> 1;          // 0..127
    const int ah = t & 1;           // 0..1
    // B staging: thread -> 4x4 register-transpose block (n0, kl)
    const int n0 = (t & 31) * 4;    // 0,4,..,124
    const int kl = (t >> 5) * 4;    // 0,4,..,28

    const int m_l = lane & 15;
    const int kh  = lane >> 4;

    const float* gA_base = F  + (size_t)(row0 + ar) * DIM + ah * 16;
    const float* gB_base = W1 + (size_t)kl * DIM + col0 + n0;

    float4 ag[4], bg[4];

    auto load_globals = [&](int k0) {
        const float4* ga = (const float4*)(gA_base + k0);
        ag[0] = ga[0]; ag[1] = ga[1]; ag[2] = ga[2]; ag[3] = ga[3];
        const float* gb = gB_base + (size_t)k0 * DIM;
        bg[0] = *(const float4*)(gb);
        bg[1] = *(const float4*)(gb + DIM);
        bg[2] = *(const float4*)(gb + 2 * DIM);
        bg[3] = *(const float4*)(gb + 3 * DIM);
    };

    auto store_tiles = [&](int buf) {
        // A tile: 16 contiguous bf16 -> two b128 stores
        uint4 p0, p1;
        p0.x = pack_bf16(ag[0].x, ag[0].y); p0.y = pack_bf16(ag[0].z, ag[0].w);
        p0.z = pack_bf16(ag[1].x, ag[1].y); p0.w = pack_bf16(ag[1].z, ag[1].w);
        p1.x = pack_bf16(ag[2].x, ag[2].y); p1.y = pack_bf16(ag[2].z, ag[2].w);
        p1.z = pack_bf16(ag[3].x, ag[3].y); p1.w = pack_bf16(ag[3].z, ag[3].w);
        uint4* da = (uint4*)&lds_a[buf][ar * BK + ah * 16];
        da[0] = p0; da[1] = p1;
        // B tile: 4x4 register transpose -> four b64 stores ([n][k] layout)
        const float* r0 = (const float*)&bg[0];
        const float* r1 = (const float*)&bg[1];
        const float* r2 = (const float*)&bg[2];
        const float* r3 = (const float*)&bg[3];
        #pragma unroll
        for (int nn = 0; nn < 4; ++nn) {
            uint2 v;
            v.x = pack_bf16(r0[nn], r1[nn]);   // k = kl+0, kl+1
            v.y = pack_bf16(r2[nn], r3[nn]);   // k = kl+2, kl+3
            *(uint2*)&lds_b[buf][(n0 + nn) * BK + kl] = v;
        }
    };

    // -------- software pipeline: prologue --------
    load_globals(0);
    store_tiles(0);
    __syncthreads();

    #pragma unroll 2
    for (int k0 = 0; k0 < DIM; k0 += BK) {
        const int  cur      = (k0 >> 5) & 1;
        const bool has_next = (k0 + BK) < DIM;

        // issue next tile's global loads early (overlap with WMMA below)
        if (has_next) load_globals(k0 + BK);

        // ---- fragment loads per ISA VGPR layouts ----
        FragU a_frag[4];
        #pragma unroll
        for (int mi = 0; mi < 4; ++mi) {
            const unsigned short* p = &lds_a[cur][(mw * 64 + mi * 16 + m_l) * BK];
            a_frag[mi].q[0] = *(const uint4*)(p + kh * 8);        // K = kh*8 .. +7
            a_frag[mi].q[1] = *(const uint4*)(p + 16 + kh * 8);   // K = 16+kh*8 .. +7
        }
        FragU b_frag[2];
        #pragma unroll
        for (int ni = 0; ni < 2; ++ni) {
            const unsigned short* p = &lds_b[cur][(nw * 32 + ni * 16 + m_l) * BK];
            b_frag[ni].q[0] = *(const uint4*)(p + kh * 8);        // K = kh*8 .. +7
            b_frag[ni].q[1] = *(const uint4*)(p + 16 + kh * 8);   // K = 16+kh*8 .. +7
        }

        // ---- 8 WMMA per K-step per wave ----
        #pragma unroll
        for (int mi = 0; mi < 4; ++mi)
            #pragma unroll
            for (int ni = 0; ni < 2; ++ni)
                acc_c[mi][ni] = __builtin_amdgcn_wmma_f32_16x16x32_bf16(
                    false, a_frag[mi].bf, false, b_frag[ni].bf,
                    (short)0, acc_c[mi][ni], false, false);

        // ---- convert + store next tile into alternate buffer ----
        if (has_next) store_tiles(cur ^ 1);
        __syncthreads();
    }

    // ---- fused epilogue: relu(+b1) * W2, reduce over this block's 128 cols ----
    if (t < BM) acc_lds[t] = 0.0f;
    __syncthreads();

    #pragma unroll
    for (int ni = 0; ni < 2; ++ni) {
        const int gcol = col0 + nw * 32 + ni * 16 + m_l;   // N = lane&15 in C layout
        const float b1v = b1[gcol];
        const float w2v = W2[gcol];
        #pragma unroll
        for (int mi = 0; mi < 4; ++mi) {
            #pragma unroll
            for (int j = 0; j < 8; ++j) {
                const int mloc = mw * 64 + mi * 16 + j + kh * 8;  // M = j (+8 hi half)
                float v = acc_c[mi][ni][j] + b1v;
                v = fmaxf(v, 0.0f) * w2v;
                atomicAdd(&acc_lds[mloc], v);
            }
        }
    }
    __syncthreads();

    if (t < BM) atomicAdd(&acc[row0 + t], acc_lds[t]);
}

// ---------------------------------------------------------------------------
// Kernel 2: per-row finalize.  scale = 1 - 0.8*sigmoid(acc+b2); top-2 of
// f*scale; faithful suppressed-winner mask; streaming masked output.
// One workgroup (256 threads) per row; 16 strided elements per thread.
// ---------------------------------------------------------------------------
__global__ __launch_bounds__(256)
void finalize_kernel(const float* __restrict__ F,
                     const float* __restrict__ acc,
                     const float* __restrict__ b2,
                     float* __restrict__ out)
{
    __shared__ float s_m1[256];
    __shared__ float s_m2[256];
    __shared__ int   s_i1[256];

    const int row = blockIdx.x;
    const int t   = threadIdx.x;

    const float z     = acc[row] + b2[0];
    const float comp  = 1.0f / (1.0f + __expf(-z));
    const float scale = 1.0f - COMP_FACTOR * comp;

    const float* frow = F + (size_t)row * DIM;

    float fv[16];
    float m1 = -INFINITY, m2 = -INFINITY;
    int   i1 = 0;
    #pragma unroll
    for (int i = 0; i < 16; ++i) {
        const int c = t + i * 256;
        const float f = frow[c];
        fv[i] = f;
        const float s = f * scale;
        if (s > m1)      { m2 = m1; m1 = s; i1 = c; }
        else if (s > m2) { m2 = s; }
    }
    s_m1[t] = m1; s_m2[t] = m2; s_i1[t] = i1;
    __syncthreads();

    for (int off = 128; off > 0; off >>= 1) {
        if (t < off) {
            const float am1 = s_m1[t],       am2 = s_m2[t];
            const float bm1 = s_m1[t + off], bm2 = s_m2[t + off];
            if (bm1 > am1) {
                s_m1[t] = bm1; s_i1[t] = s_i1[t + off];
                s_m2[t] = fmaxf(bm2, am1);
            } else {
                s_m2[t] = fmaxf(am2, bm1);
            }
        }
        __syncthreads();
    }
    m1 = s_m1[0]; m2 = s_m2[0]; i1 = s_i1[0];

    float* orow = out + (size_t)row * DIM;
    #pragma unroll
    for (int i = 0; i < 16; ++i) {
        const int c = t + i * 256;
        const float f = fv[i];
        const float excl  = (c == i1) ? m2 : m1;
        const float other = fmaxf(excl, f);
        const bool  win   = other < f;     // identically false by algebra; kept faithful
        orow[c] = (f > THRESHOLD) ? (win ? f : 0.0f) : f;
    }
}

// ---------------------------------------------------------------------------
// Launch
// ---------------------------------------------------------------------------
extern "C" void kernel_launch(void* const* d_in, const int* in_sizes, int n_in,
                              void* d_out, int out_size, void* d_ws, size_t ws_size,
                              hipStream_t stream) {
    const float* F  = (const float*)d_in[0];
    const float* W1 = (const float*)d_in[1];
    const float* b1 = (const float*)d_in[2];
    const float* W2 = (const float*)d_in[3];
    const float* b2 = (const float*)d_in[4];
    float* out = (float*)d_out;
    float* acc = (float*)d_ws;                 // BATCH floats = 16 KB scratch

    zero_acc_kernel<<<(BATCH + 255) / 256, 256, 0, stream>>>(acc, BATCH);

    dim3 grid(DIM / BN, BATCH / BM);           // 32 x 32 workgroups
    gemm_comp_kernel<<<grid, 256, 0, stream>>>(F, W1, b1, W2, acc);

    finalize_kernel<<<BATCH, 256, 0, stream>>>(F, acc, b2, out);
}
